// MultiScaleDeformAttn_34505767256646
// MI455X (gfx1250) — matrix operand order repaired and checked by reading, not compiled, not run on hardware
//
#include <hip/hip_runtime.h>
#include <hip/hip_bf16.h>
#include <math.h>

typedef __bf16 bf16_t;
typedef __attribute__((ext_vector_type(16))) __bf16 v16bf;
typedef __attribute__((ext_vector_type(8)))  __bf16 v8bf;
typedef __attribute__((ext_vector_type(8)))  float  v8f;

// ---- problem constants (from reference setup) ----
#define NBATCH 2
#define QLEN   11253
#define SLEN   11253
#define CDIM   256
#define NQ_TOT (NBATCH * QLEN)   // 22506 rows for all GEMMs
#define HHEADS 8
#define NLVL   4
#define NPTS   4
#define DDIM   32

// ---------------- fp32 -> bf16 weight conversion ----------------
__global__ void f32_to_bf16_kernel(const float* __restrict__ src,
                                   bf16_t* __restrict__ dst, int n) {
    int i = blockIdx.x * blockDim.x + threadIdx.x;
    if (i < n) dst[i] = (bf16_t)src[i];
}

// ---------------- bf16 WMMA GEMM ----------------
// Y[M x (NT*16)] = X[M x 256] @ W^T + bias
//   X: fp32 (converted to bf16 in registers), W: bf16 row-major (NT*16, 256)
// One wave computes a 16-row tile across the full output width.
template<int NT, typename OutT>
__global__ void __launch_bounds__(256)
gemm_bf16_wmma(const float* __restrict__ X,
               const bf16_t* __restrict__ W,
               const float* __restrict__ bias,
               OutT* __restrict__ out,
               int Mtotal) {
    const int lane = threadIdx.x & 31;
    const int wave = (int)((blockIdx.x * blockDim.x + threadIdx.x) >> 5);
    const int l16  = lane & 15;
    const int half = lane >> 4;
    const int mt   = wave;                 // 16-row tile index
    if (mt * 16 >= Mtotal) return;

    const int  row   = mt * 16 + l16;      // A operand: M = lane%16
    const bool rowok = row < Mtotal;

    v8f acc[NT];
#pragma unroll
    for (int nt = 0; nt < NT; ++nt) acc[nt] = {};

    const v8f zf = {};
    union AU { v16bf v; v8bf h[2]; };

#pragma unroll
    for (int kt = 0; kt < CDIM / 32; ++kt) {
        // A fragment: lane holds K = kt*32 + 16*half + 0..15
        const int kbase = kt * 32 + half * 16;
        AU a;
        if (rowok) {
            v8f lo = *(const v8f*)(X + (size_t)row * CDIM + kbase);
            v8f hi = *(const v8f*)(X + (size_t)row * CDIM + kbase + 8);
            a.h[0] = __builtin_convertvector(lo, v8bf);
            a.h[1] = __builtin_convertvector(hi, v8bf);
        } else {
            a.h[0] = __builtin_convertvector(zf, v8bf);
            a.h[1] = __builtin_convertvector(zf, v8bf);
        }
#pragma unroll
        for (int nt = 0; nt < NT; ++nt) {
            // B fragment: lane holds column n = nt*16 + lane%16,
            // K = kt*32 + 16*half + 0..15  -> 16 contiguous bf16 of W row n
            const int ncol = nt * 16 + l16;
            v16bf b = *(const v16bf*)(W + (size_t)ncol * CDIM + kt * 32 + half * 16);
            acc[nt] = __builtin_amdgcn_wmma_f32_16x16x32_bf16(
                false, a.v, false, b, (short)0, acc[nt], false, false);
        }
    }

    // C/D layout: VGPR r, lane: N = lane%16, M = r + 8*(lane/16)
    const int Ncols = NT * 16;
#pragma unroll
    for (int nt = 0; nt < NT; ++nt) {
        const int ncol = nt * 16 + l16;
        const float bv = bias[ncol];
#pragma unroll
        for (int r = 0; r < 8; ++r) {
            const int m = mt * 16 + r + 8 * half;
            if (m < Mtotal)
                out[(size_t)m * Ncols + ncol] = (OutT)(acc[nt][r] + bv);
        }
    }
}

// ---------------- deformable bilinear gather ----------------
// one wave per (n, q, h); lane = channel d in [0, 32)
__global__ void __launch_bounds__(256)
deform_gather_kernel(const float* __restrict__ p,
                     const float* __restrict__ attn,
                     const float* __restrict__ locoff,
                     const bf16_t* __restrict__ vproj,
                     const int* __restrict__ shapes,
                     const int* __restrict__ level_index,
                     float* __restrict__ accout) {
    const int lane = threadIdx.x & 31;
    const int wave = (int)((blockIdx.x * blockDim.x + threadIdx.x) >> 5);
    if (wave >= NQ_TOT * HHEADS) return;
    const int h  = wave & (HHEADS - 1);
    const int nq = wave >> 3;
    const int n  = nq / QLEN;

    const int S = level_index[NLVL - 1] +
                  shapes[2 * (NLVL - 1)] * shapes[2 * (NLVL - 1) + 1];

    float acc = 0.0f;
#pragma unroll
    for (int l = 0; l < NLVL; ++l) {
        const int   Hl = shapes[2 * l];
        const int   Wl = shapes[2 * l + 1];
        const int   st = level_index[l];
        const float px = p[(size_t)nq * (NLVL * 2) + l * 2 + 0];
        const float py = p[(size_t)nq * (NLVL * 2) + l * 2 + 1];
#pragma unroll
        for (int pt = 0; pt < NPTS; ++pt) {
            const int   fidx = (h * NLVL + l) * NPTS + pt;
            const float aw = attn[(size_t)nq * (HHEADS * NLVL * NPTS) + fidx];
            const float ox = locoff[(size_t)nq * (HHEADS * NLVL * NPTS * 2) + fidx * 2 + 0];
            const float oy = locoff[(size_t)nq * (HHEADS * NLVL * NPTS * 2) + fidx * 2 + 1];
            // loc = p + off/norm  =>  x = loc.x*W - 0.5 = px*W + ox - 0.5
            const float x = px * (float)Wl + ox - 0.5f;
            const float y = py * (float)Hl + oy - 0.5f;
            const float x0f = floorf(x), y0f = floorf(y);
            const float lx = x - x0f, ly = y - y0f;
            const int   x0 = (int)x0f, y0 = (int)y0f;
            const float ws[4] = { (1.f - ly) * (1.f - lx), (1.f - ly) * lx,
                                  ly * (1.f - lx),         ly * lx };
            const int   xs[4] = { x0, x0 + 1, x0,     x0 + 1 };
            const int   ys[4] = { y0, y0,     y0 + 1, y0 + 1 };
#pragma unroll
            for (int c = 0; c < 4; ++c) {
                const int  xi = xs[c], yi = ys[c];
                const bool valid = (xi >= 0) & (xi < Wl) & (yi >= 0) & (yi < Hl);
                int idx = st + yi * Wl + xi;
                idx = idx < 0 ? 0 : (idx > S - 1 ? S - 1 : idx);
                const float g = (float)vproj[((size_t)n * SLEN + idx) * CDIM + h * DDIM + lane];
                acc += valid ? aw * ws[c] * g : 0.0f;
            }
        }
    }
    accout[(size_t)nq * CDIM + h * DDIM + lane] = acc;
}

extern "C" void kernel_launch(void* const* d_in, const int* in_sizes, int n_in,
                              void* d_out, int out_size, void* d_ws, size_t ws_size,
                              hipStream_t stream) {
    const float* q      = (const float*)d_in[0];
    const float* p      = (const float*)d_in[1];
    const float* v      = (const float*)d_in[2];
    const int*   shapes = (const int*)d_in[3];
    const int*   lvl    = (const int*)d_in[4];
    const float* W_off  = (const float*)d_in[5];
    const float* b_off  = (const float*)d_in[6];
    const float* W_attn = (const float*)d_in[7];
    const float* b_attn = (const float*)d_in[8];
    const float* W_v    = (const float*)d_in[9];
    const float* b_v    = (const float*)d_in[10];
    const float* W_out  = (const float*)d_in[11];
    const float* b_out  = (const float*)d_in[12];
    (void)in_sizes; (void)n_in; (void)out_size; (void)ws_size;

    // workspace carving (256B aligned)
    char*  ws  = (char*)d_ws;
    size_t off = 0;
    auto alloc = [&](size_t bytes) -> char* {
        char* ptr = ws + off;
        off += (bytes + 255) & ~(size_t)255;
        return ptr;
    };
    bf16_t* Wv_bf  = (bf16_t*)alloc((size_t)CDIM * CDIM * sizeof(bf16_t));
    bf16_t* Wat_bf = (bf16_t*)alloc((size_t)128  * CDIM * sizeof(bf16_t));
    bf16_t* Wof_bf = (bf16_t*)alloc((size_t)CDIM * CDIM * sizeof(bf16_t));
    bf16_t* Wou_bf = (bf16_t*)alloc((size_t)CDIM * CDIM * sizeof(bf16_t));
    bf16_t* vproj  = (bf16_t*)alloc((size_t)NQ_TOT * CDIM * sizeof(bf16_t)); // 11.5 MB, L2-resident
    float*  attnb  = (float*) alloc((size_t)NQ_TOT * 128  * sizeof(float));
    float*  locoff = (float*) alloc((size_t)NQ_TOT * CDIM * sizeof(float));
    float*  accbuf = (float*) alloc((size_t)NQ_TOT * CDIM * sizeof(float));

    // 1. weights -> bf16 (tiny)
    f32_to_bf16_kernel<<<(CDIM * CDIM + 255) / 256, 256, 0, stream>>>(W_v,    Wv_bf,  CDIM * CDIM);
    f32_to_bf16_kernel<<<(128  * CDIM + 255) / 256, 256, 0, stream>>>(W_attn, Wat_bf, 128 * CDIM);
    f32_to_bf16_kernel<<<(CDIM * CDIM + 255) / 256, 256, 0, stream>>>(W_off,  Wof_bf, CDIM * CDIM);
    f32_to_bf16_kernel<<<(CDIM * CDIM + 255) / 256, 256, 0, stream>>>(W_out,  Wou_bf, CDIM * CDIM);

    const int Mtiles   = (NQ_TOT + 15) / 16;   // 1407 16-row tiles
    const int gemmGrid = (Mtiles + 7) / 8;     // 8 waves per 256-thread block

    // 2. projections (bf16 WMMA, fp32 accumulate)
    gemm_bf16_wmma<16, bf16_t><<<gemmGrid, 256, 0, stream>>>(v, Wv_bf,  b_v,    vproj,  NQ_TOT);
    gemm_bf16_wmma<8,  float ><<<gemmGrid, 256, 0, stream>>>(q, Wat_bf, b_attn, attnb,  NQ_TOT);
    gemm_bf16_wmma<16, float ><<<gemmGrid, 256, 0, stream>>>(q, Wof_bf, b_off,  locoff, NQ_TOT);

    // 3. deformable bilinear gather (fp32 math, bf16 L2-resident value reads)
    deform_gather_kernel<<<NQ_TOT, 256, 0, stream>>>(p, attnb, locoff, vproj, shapes, lvl, accbuf);

    // 4. output projection
    gemm_bf16_wmma<16, float ><<<gemmGrid, 256, 0, stream>>>(accbuf, Wou_bf, b_out, (float*)d_out, NQ_TOT);
}
